// STTransformerBlock_24635932410451
// MI455X (gfx1250) — matrix operand order repaired and checked by reading, not compiled, not run on hardware
//
#include <hip/hip_runtime.h>
#include <hip/hip_bf16.h>
#include <cstdint>

// ---------------------------------------------------------------------------
// ST transformer block for MI455X (gfx1250, wave32, WMMA + TDM).
// GEMMs: bf16 x bf16 -> fp32 via v_wmma_f32_16x16x32_bf16, B panel staged in
// LDS by the Tensor Data Mover (double-buffered, TENSORcnt-synchronized).
// Softmax / LayerNorm / masking stay in fp32 VALU.
// ---------------------------------------------------------------------------

typedef __attribute__((ext_vector_type(16))) __bf16 v16bf;
typedef __attribute__((ext_vector_type(8)))  __bf16 v8bf;
typedef __attribute__((ext_vector_type(8)))  float  v8f;
typedef __attribute__((ext_vector_type(4)))  unsigned int v4u;
typedef __attribute__((ext_vector_type(8)))  int v8i_;
typedef __attribute__((ext_vector_type(4)))  int v4i_;

#define CB  4      // batch
#define CS  16     // seq (temporal)
#define CP  256    // patches (spatial)
#define CE  768    // embed
#define CH  12     // heads
#define CHD 64     // head dim
#define CFF 3072   // ffn hidden
#define CNT (CB*CS*CP)   // 16384 tokens
#define KC  128    // TDM K-chunk (bf16 elements)

__device__ __forceinline__ v8f vzero8() {
  v8f z;
#pragma unroll
  for (int i = 0; i < 8; ++i) z[i] = 0.f;
  return z;
}

// Build the 16-bit A-operand (16x32 tile slice for this lane) from two
// contiguous 8-half chunks: elements 0..15 = K[koff..koff+15]
// (contiguous order, matching the B operand).
__device__ __forceinline__ v16bf load_a16(const __bf16* p) {
  v8bf lo = *(const v8bf*)(p);
  v8bf hi = *(const v8bf*)(p + 8);
  v16bf a;
#pragma unroll
  for (int i = 0; i < 8; ++i) { a[i] = lo[i]; a[i + 8] = hi[i]; }
  return a;
}

__device__ __forceinline__ float redmax16(float v) {
#pragma unroll
  for (int m = 1; m < 16; m <<= 1) v = fmaxf(v, __shfl_xor(v, m, 32));
  return v;
}
__device__ __forceinline__ float redsum16(float v) {
#pragma unroll
  for (int m = 1; m < 16; m <<= 1) v += __shfl_xor(v, m, 32);
  return v;
}

// ---------------------------------------------------------------------------
// TDM: DMA a [rows x KC] bf16 tile (row stride = K elements) from global
// memory into LDS at byte offset lds_byte_off. Descriptor packed per CDNA5
// ISA 8.3/8.4 (D# group0 + group1; 2-D tensor so groups 2/3 are zero).
// This toolchain exposes the 6-arg builtin (g0, g1, g2, g3, g_extra, cpol).
// ---------------------------------------------------------------------------
__device__ __forceinline__ void tdm_load_b_panel(const __bf16* gsrc,
                                                 unsigned lds_byte_off,
                                                 int K, int rows) {
  const unsigned long long ga = (unsigned long long)(uintptr_t)gsrc;
  v4u g0;
  g0[0] = 1u;                                            // count=1 (valid, user)
  g0[1] = lds_byte_off;                                  // lds_addr
  g0[2] = (unsigned)(ga & 0xFFFFFFFFu);                  // global_addr[31:0]
  g0[3] = (unsigned)((ga >> 32) & 0x1FFFFFFu)            // global_addr[56:32]
          | (2u << 30);                                  // type=2 ("image")
  const unsigned td0 = (unsigned)K;                      // tensor_dim0 (elems)
  const unsigned td1 = (unsigned)rows;                   // tensor_dim1
  v8i_ g1;
  g1[0] = (int)(1u << 16);                               // data_size=1 -> 2B
  g1[1] = (int)((td0 & 0xFFFFu) << 16);                  // tensor_dim0[15:0]
  g1[2] = (int)(((td0 >> 16) & 0xFFFFu) |
                ((td1 & 0xFFFFu) << 16));                // td0[31:16], td1[15:0]
  g1[3] = (int)(((td1 >> 16) & 0xFFFFu) |
                ((unsigned)KC << 16));                   // td1[31:16], tile_dim0
  g1[4] = (int)((unsigned)rows & 0xFFFFu);               // tile_dim1 (tile_dim2=0)
  g1[5] = (int)(unsigned)K;                              // tensor_dim0_stride lo
  g1[6] = 0;                                             // stride0 hi, stride1 lo
  g1[7] = 0;
  v4i_ gz4;
  gz4[0] = 0; gz4[1] = 0; gz4[2] = 0; gz4[3] = 0;
  v8i_ gz8;
#pragma unroll
  for (int i = 0; i < 8; ++i) gz8[i] = 0;
  __builtin_amdgcn_tensor_load_to_lds(g0, g1, gz4, gz4, gz8, 0);
}

// ---------------------------------------------------------------------------
// fp32 -> bf16 elementwise convert
// ---------------------------------------------------------------------------
__global__ void cvt_f32_bf16_kernel(const float* __restrict__ s,
                                    __bf16* __restrict__ d, size_t n) {
  size_t i = (size_t)blockIdx.x * 256 + threadIdx.x;
  if (i < n) d[i] = (__bf16)s[i];
}

// ---------------------------------------------------------------------------
// Transpose + convert weights: W[K][N] fp32 -> Wt[N][K] bf16 (LDS tiled)
// ---------------------------------------------------------------------------
__global__ __launch_bounds__(256)
void transpose_cvt_kernel(const float* __restrict__ W, __bf16* __restrict__ Wt,
                          int K, int N) {
  __shared__ float tile[32][33];
  const int kt = blockIdx.y * 32, nt = blockIdx.x * 32;
  const int tx = threadIdx.x & 31, ty = threadIdx.x >> 5;
#pragma unroll
  for (int r = ty; r < 32; r += 8) {
    int k = kt + r, n = nt + tx;
    tile[r][tx] = (k < K && n < N) ? W[(size_t)k * N + n] : 0.f;
  }
  __syncthreads();
#pragma unroll
  for (int r = ty; r < 32; r += 8) {
    int n = nt + r, k = kt + tx;
    if (n < N && k < K) Wt[(size_t)n * K + k] = (__bf16)tile[tx][r];
  }
}

// ---------------------------------------------------------------------------
// GEMM: C[M,N] = A[M,K](bf16, row-major) x Bt[N,K](bf16) + bias, fused
// relu / fp32 store / bf16 store. Wave = 16(M) x 64(N), block = 8 waves
// covering 128(M) x 64(N). The 64xK B panel is staged through LDS by the
// TDM in KC-element chunks, double-buffered, shared by all 8 waves.
// Inner loop batches all 4 B-tile LDS loads per k-step so the XDL pipe can
// take the 4 WMMAs back-to-back behind a single DScnt wait.
// grid = (N/64, M/128); dynamic LDS = 2 * 64 * KC * 2 bytes = 32 KB.
// ---------------------------------------------------------------------------
__global__ __launch_bounds__(256)
void gemm_bf16_tdm_kernel(const __bf16* __restrict__ A,
                          const __bf16* __restrict__ Bt,
                          const float* __restrict__ bias,
                          float* __restrict__ Cf, __bf16* __restrict__ Cb,
                          int M, int N, int K, int relu) {
  extern __shared__ __bf16 sB[];           // 2 x [64][KC]
  const int lane  = threadIdx.x & 31;
  const int wave  = threadIdx.x >> 5;
  const int m0    = blockIdx.y * 128 + wave * 16;
  const int n0    = blockIdx.x * 64;
  const int mrow  = lane & 15;
  const int hi    = (lane >> 4) & 1;
  const int koffA = hi ? 16 : 0;
  const int koffB = hi ? 16 : 0;

  v8f acc[4];
#pragma unroll
  for (int t = 0; t < 4; ++t) acc[t] = vzero8();

  const __bf16* Arow   = A  + (size_t)(m0 + mrow) * K + koffA;
  const __bf16* Bpanel = Bt + (size_t)n0 * K;
  const int nch = K / KC;
  const unsigned bufBytes = 64u * KC * sizeof(__bf16);   // 16 KB

  if (wave == 0) tdm_load_b_panel(Bpanel, 0u, K, 64);

  for (int c = 0; c < nch; ++c) {
    if (wave == 0) __builtin_amdgcn_s_wait_tensorcnt(0); // chunk c landed
    __syncthreads();                                     // publish + reuse-safe
    if (wave == 0 && c + 1 < nch)
      tdm_load_b_panel(Bpanel + (size_t)(c + 1) * KC,
                       (unsigned)((c + 1) & 1) * bufBytes, K, 64);

    const __bf16* sbuf = sB + (size_t)(c & 1) * 64 * KC + (size_t)mrow * KC + koffB;
    const int kbase = c * KC;
#pragma unroll
    for (int kk = 0; kk < KC; kk += 32) {
      v16bf a = load_a16(Arow + kbase + kk);
      v16bf b0 = *(const v16bf*)(sbuf + 0 * 16 * KC + kk);
      v16bf b1 = *(const v16bf*)(sbuf + 1 * 16 * KC + kk);
      v16bf b2 = *(const v16bf*)(sbuf + 2 * 16 * KC + kk);
      v16bf b3 = *(const v16bf*)(sbuf + 3 * 16 * KC + kk);
      acc[0] = __builtin_amdgcn_wmma_f32_16x16x32_bf16(
          false, a, false, b0, (short)0, acc[0], false, false);
      acc[1] = __builtin_amdgcn_wmma_f32_16x16x32_bf16(
          false, a, false, b1, (short)0, acc[1], false, false);
      acc[2] = __builtin_amdgcn_wmma_f32_16x16x32_bf16(
          false, a, false, b2, (short)0, acc[2], false, false);
      acc[3] = __builtin_amdgcn_wmma_f32_16x16x32_bf16(
          false, a, false, b3, (short)0, acc[3], false, false);
    }
  }

#pragma unroll
  for (int t = 0; t < 4; ++t) {
    const int col = n0 + t * 16 + mrow;
    const float bv = bias ? bias[col] : 0.f;
#pragma unroll
    for (int i = 0; i < 8; ++i) {
      const int row = m0 + i + hi * 8;
      float v = acc[t][i] + bv;
      if (relu) v = fmaxf(v, 0.f);
      const size_t idx = (size_t)row * N + col;
      if (Cf) Cf[idx] = v;
      if (Cb) Cb[idx] = (__bf16)v;
    }
  }
}

// ---------------------------------------------------------------------------
// V fp32 [bs, p, h, d] -> Vt bf16 [(bs*H+h), d, p]  (B-operand for attn*V)
// ---------------------------------------------------------------------------
__global__ void vtrans_spatial_kernel(const float* __restrict__ vf,
                                      __bf16* __restrict__ vt) {
  size_t o = (size_t)blockIdx.x * 256 + threadIdx.x;   // total = 768*64*256
  const int p  = (int)(o & 255);
  const int d  = (int)((o >> 8) & 63);
  const int g  = (int)(o >> 14);
  const int h  = g % CH;
  const int bs = g / CH;
  vt[o] = (__bf16)vf[((size_t)bs * CP + p) * CE + h * CHD + d];
}

// ---------------------------------------------------------------------------
// Spatial attention, fused scores->softmax->AV per (b,s,h) group.
// block = 8 waves, each wave owns 2 query tiles of 16 rows.
// ---------------------------------------------------------------------------
__global__ __launch_bounds__(256)
void spatial_attn_kernel(const __bf16* __restrict__ qb,
                         const __bf16* __restrict__ kb,
                         const __bf16* __restrict__ vtb,
                         __bf16* __restrict__ attb) {
  __shared__ __align__(64) __bf16 sattn[8][16 * CP];   // 64 KB
  const int g    = blockIdx.x;           // (b*S+s)*H + h
  const int h    = g % CH;
  const int bs   = g / CH;
  const int lane = threadIdx.x & 31;
  const int wave = threadIdx.x >> 5;
  const int mrow = lane & 15;
  const int hi   = (lane >> 4) & 1;
  const int koffA = hi ? 16 : 0;
  const int koffB = hi ? 16 : 0;
  const float scale = 0.125f;            // 1/sqrt(64)

  const __bf16* qg = qb  + (size_t)bs * CP * CE + h * CHD;
  const __bf16* kg = kb  + (size_t)bs * CP * CE + h * CHD;
  const __bf16* vg = vtb + (size_t)g * CHD * CP;

  for (int ti = wave * 2; ti < wave * 2 + 2; ++ti) {
    // ---- scores: 16 x 256 strip, K = head-dim (64) ----
    v8f acc[16];
#pragma unroll
    for (int tj = 0; tj < 16; ++tj) acc[tj] = vzero8();

    const __bf16* arow = qg + (size_t)(ti * 16 + mrow) * CE + koffA;
#pragma unroll
    for (int kk = 0; kk < CHD; kk += 32) {
      v16bf a = load_a16(arow + kk);
#pragma unroll
      for (int tj = 0; tj < 16; ++tj) {
        v16bf b = *(const v16bf*)(kg + (size_t)(tj * 16 + mrow) * CE + koffB + kk);
        acc[tj] = __builtin_amdgcn_wmma_f32_16x16x32_bf16(
            false, a, false, b, (short)0, acc[tj], false, false);
      }
    }

    // ---- softmax over 256 keys per query row ----
    float rmax[8], rinv[8];
#pragma unroll
    for (int i = 0; i < 8; ++i) {
      float m = -3.0e38f;
#pragma unroll
      for (int tj = 0; tj < 16; ++tj) m = fmaxf(m, acc[tj][i]);
      rmax[i] = redmax16(m);
    }
#pragma unroll
    for (int i = 0; i < 8; ++i) {
      float s = 0.f;
#pragma unroll
      for (int tj = 0; tj < 16; ++tj) {
        float e = __expf((acc[tj][i] - rmax[i]) * scale);
        acc[tj][i] = e;
        s += e;
      }
      s = redsum16(s);
      rinv[i] = 1.f / s;
    }

    // ---- C-layout -> row-major bf16 attn in LDS ----
#pragma unroll
    for (int tj = 0; tj < 16; ++tj)
#pragma unroll
      for (int i = 0; i < 8; ++i) {
        const int r = i + hi * 8;
        sattn[wave][r * CP + tj * 16 + mrow] = (__bf16)(acc[tj][i] * rinv[i]);
      }
    asm volatile("s_wait_dscnt 0" ::: "memory");

    // ---- out = attn x V : 16 x 64, K = 256 (A from LDS, B = Vt) ----
    v8f oacc[4];
#pragma unroll
    for (int t = 0; t < 4; ++t) oacc[t] = vzero8();
    const __bf16* sa = &sattn[wave][mrow * CP + koffA];
    for (int k = 0; k < CP; k += 32) {
      v16bf a = load_a16(sa + k);
#pragma unroll
      for (int t = 0; t < 4; ++t) {
        v16bf b = *(const v16bf*)(vg + (size_t)(t * 16 + mrow) * CP + koffB + k);
        oacc[t] = __builtin_amdgcn_wmma_f32_16x16x32_bf16(
            false, a, false, b, (short)0, oacc[t], false, false);
      }
    }
#pragma unroll
    for (int t = 0; t < 4; ++t) {
      const int col = h * CHD + t * 16 + mrow;
#pragma unroll
      for (int i = 0; i < 8; ++i) {
        const int tok = ti * 16 + i + hi * 8;
        attb[((size_t)bs * CP + tok) * CE + col] = (__bf16)oacc[t][i];
      }
    }
  }
}

// ---------------------------------------------------------------------------
// Temporal (causal) attention: one wave per (b,p,h) group, S=16.
// Scores via WMMA (K=64), softmax fp32, tiny attn*V in VALU via LDS.
// ---------------------------------------------------------------------------
__global__ __launch_bounds__(256)
void temporal_attn_kernel(const __bf16* __restrict__ qb,
                          const __bf16* __restrict__ kb,
                          const float* __restrict__ vf,
                          __bf16* __restrict__ attb) {
  __shared__ float sat[8][16 * 16];
  const int lane = threadIdx.x & 31;
  const int wave = threadIdx.x >> 5;
  const int gid  = blockIdx.x * 8 + wave;          // in [0, B*P*H)
  const int h  = gid % CH;
  const int bp = gid / CH;
  const int p  = bp % CP;
  const int b  = bp / CP;
  const int mrow = lane & 15;
  const int hi   = (lane >> 4) & 1;
  const int koffA = hi ? 16 : 0;
  const int koffB = hi ? 16 : 0;
  const float scale = 0.125f;
  const size_t rowstride = (size_t)CP * CE;        // stride over s

  const __bf16* qg = qb + ((size_t)b * CS * CP + p) * CE + h * CHD;
  const __bf16* kg = kb + ((size_t)b * CS * CP + p) * CE + h * CHD;

  v8f acc = vzero8();
#pragma unroll
  for (int kk = 0; kk < CHD; kk += 32) {
    v16bf a = load_a16(qg + (size_t)mrow * rowstride + koffA + kk);
    v16bf bmat = *(const v16bf*)(kg + (size_t)mrow * rowstride + koffB + kk);
    acc = __builtin_amdgcn_wmma_f32_16x16x32_bf16(
        false, a, false, bmat, (short)0, acc, false, false);
  }

  // causal mask + softmax (row m = i + hi*8, col = mrow)
  float vals[8];
#pragma unroll
  for (int i = 0; i < 8; ++i) {
    const int row = i + hi * 8;
    vals[i] = (mrow <= row) ? acc[i] * scale : -3.0e38f;
  }
#pragma unroll
  for (int i = 0; i < 8; ++i) {
    float m = redmax16(vals[i]);
    float e = __expf(vals[i] - m);
    float s = redsum16(e);
    const int row = i + hi * 8;
    sat[wave][row * 16 + mrow] = e / s;
  }
  asm volatile("s_wait_dscnt 0" ::: "memory");

  // out(m, d) = sum_s attn(m,s) * V(s,d); lane covers d and d+32
  for (int m = 0; m < 16; ++m) {
    float o0 = 0.f, o1 = 0.f;
    for (int s = 0; s <= m; ++s) {
      const float a = sat[wave][m * 16 + s];
      const float* vrow = vf + (((size_t)b * CS + s) * CP + p) * CE + h * CHD;
      o0 += a * vrow[lane];
      o1 += a * vrow[lane + 32];
    }
    const size_t idx = (((size_t)b * CS + m) * CP + p) * CE + h * CHD;
    attb[idx + lane]      = (__bf16)o0;
    attb[idx + lane + 32] = (__bf16)o1;
  }
}

// ---------------------------------------------------------------------------
// Fused residual-add + LayerNorm over E=768; emits fp32 x' and bf16 x'.
// block = 256 threads per token row, each thread owns 3 columns.
// ---------------------------------------------------------------------------
__global__ __launch_bounds__(256)
void add_ln_kernel(const float* __restrict__ xres, const float* __restrict__ y,
                   const float* __restrict__ gg, const float* __restrict__ bb,
                   float* __restrict__ outf, __bf16* __restrict__ outb) {
  const int row  = blockIdx.x;
  const int tid  = threadIdx.x;
  const int lane = tid & 31, wave = tid >> 5;
  const float* xr = xres + (size_t)row * CE;
  const float* yr = y    + (size_t)row * CE;
  __shared__ float red[8];
  __shared__ float stat[2];

  float v[3];
  float s = 0.f;
#pragma unroll
  for (int j = 0; j < 3; ++j) {
    const int c = tid + j * 256;
    const float t = xr[c] + yr[c];
    v[j] = t;
    s += t;
  }
#pragma unroll
  for (int m = 1; m < 32; m <<= 1) s += __shfl_xor(s, m, 32);
  if (lane == 0) red[wave] = s;
  __syncthreads();
  if (tid == 0) {
    float t = 0.f;
    for (int w = 0; w < 8; ++w) t += red[w];
    stat[0] = t * (1.f / CE);
  }
  __syncthreads();
  const float mean = stat[0];

  float q = 0.f;
#pragma unroll
  for (int j = 0; j < 3; ++j) { const float d = v[j] - mean; q += d * d; }
#pragma unroll
  for (int m = 1; m < 32; m <<= 1) q += __shfl_xor(q, m, 32);
  if (lane == 0) red[wave] = q;
  __syncthreads();
  if (tid == 0) {
    float t = 0.f;
    for (int w = 0; w < 8; ++w) t += red[w];
    stat[1] = rsqrtf(t * (1.f / CE) + 1e-5f);
  }
  __syncthreads();
  const float inv = stat[1];

#pragma unroll
  for (int j = 0; j < 3; ++j) {
    const int c = tid + j * 256;
    const float o = (v[j] - mean) * inv * gg[c] + bb[c];
    const size_t idx = (size_t)row * CE + c;
    outf[idx] = o;
    if (outb) outb[idx] = (__bf16)o;
  }
}

// ---------------------------------------------------------------------------
// Host-side orchestration
// ---------------------------------------------------------------------------
extern "C" void kernel_launch(void* const* d_in, const int* in_sizes, int n_in,
                              void* d_out, int out_size, void* d_ws, size_t ws_size,
                              hipStream_t stream) {
  const float* x        = (const float*)d_in[0];
  const float* sa_wqkv  = (const float*)d_in[1];
  const float* sa_bqkv  = (const float*)d_in[2];
  const float* sa_wo    = (const float*)d_in[3];
  const float* sa_bo    = (const float*)d_in[4];
  const float* sa_g     = (const float*)d_in[5];
  const float* sa_b     = (const float*)d_in[6];
  const float* ta_wqkv  = (const float*)d_in[7];
  const float* ta_bqkv  = (const float*)d_in[8];
  const float* ta_wo    = (const float*)d_in[9];
  const float* ta_bo    = (const float*)d_in[10];
  const float* ta_g     = (const float*)d_in[11];
  const float* ta_b     = (const float*)d_in[12];
  const float* f_w1     = (const float*)d_in[13];
  const float* f_b1     = (const float*)d_in[14];
  const float* f_w2     = (const float*)d_in[15];
  const float* f_b2     = (const float*)d_in[16];
  const float* f_g      = (const float*)d_in[17];
  const float* f_b      = (const float*)d_in[18];
  float* out = (float*)d_out;

  const size_t EE = (size_t)CE * CE;
  char* wsp = (char*)d_ws;
  auto alloc = [&](size_t bytes) -> void* {
    void* p = (void*)wsp;
    wsp += (bytes + 255) & ~(size_t)255;
    return p;
  };

  __bf16* w_saqkv_t = (__bf16*)alloc(3 * EE * sizeof(__bf16));
  __bf16* w_sawo_t  = (__bf16*)alloc(EE * sizeof(__bf16));
  __bf16* w_taqkv_t = (__bf16*)alloc(3 * EE * sizeof(__bf16));
  __bf16* w_tawo_t  = (__bf16*)alloc(EE * sizeof(__bf16));
  __bf16* w_f1_t    = (__bf16*)alloc((size_t)CE * CFF * sizeof(__bf16)); // [FF][E]
  __bf16* w_f2_t    = (__bf16*)alloc((size_t)CE * CFF * sizeof(__bf16)); // [E][FF]

  __bf16* xb   = (__bf16*)alloc((size_t)CNT * CE * sizeof(__bf16));
  __bf16* qbf  = (__bf16*)alloc((size_t)CNT * CE * sizeof(__bf16));
  __bf16* kbf  = (__bf16*)alloc((size_t)CNT * CE * sizeof(__bf16));
  float*  vf   = (float*) alloc((size_t)CNT * CE * sizeof(float));
  __bf16* vtb  = (__bf16*)alloc((size_t)CNT * CE * sizeof(__bf16));
  __bf16* attb = (__bf16*)alloc((size_t)CNT * CE * sizeof(__bf16));
  float*  proj = (float*) alloc((size_t)CNT * CE * sizeof(float));
  float*  xA   = (float*) alloc((size_t)CNT * CE * sizeof(float));
  float*  xB   = (float*) alloc((size_t)CNT * CE * sizeof(float));
  __bf16* hb   = (__bf16*)alloc((size_t)CNT * CFF * sizeof(__bf16));

  const dim3 blk(256);
  const dim3 gT(CE / 32, CE / 32);          // E x E transpose tiles
  const size_t nXE = (size_t)CNT * CE;
  const unsigned gemmLds = 2u * 64u * KC * sizeof(__bf16);   // 32 KB

  // ---- weight prep (transpose + bf16) ----
  for (int t = 0; t < 3; ++t) {
    transpose_cvt_kernel<<<gT, blk, 0, stream>>>(sa_wqkv + t * EE, w_saqkv_t + t * EE, CE, CE);
    transpose_cvt_kernel<<<gT, blk, 0, stream>>>(ta_wqkv + t * EE, w_taqkv_t + t * EE, CE, CE);
  }
  transpose_cvt_kernel<<<gT, blk, 0, stream>>>(sa_wo, w_sawo_t, CE, CE);
  transpose_cvt_kernel<<<gT, blk, 0, stream>>>(ta_wo, w_tawo_t, CE, CE);
  transpose_cvt_kernel<<<dim3(CFF / 32, CE / 32), blk, 0, stream>>>(f_w1, w_f1_t, CE, CFF);
  transpose_cvt_kernel<<<dim3(CE / 32, CFF / 32), blk, 0, stream>>>(f_w2, w_f2_t, CFF, CE);

  const dim3 gGE(CE / 64, CNT / 128);    // GEMM N=768
  const dim3 gGF(CFF / 64, CNT / 128);   // GEMM N=3072

  // ================= stage 1: spatial attention =================
  cvt_f32_bf16_kernel<<<dim3((unsigned)((nXE + 255) / 256)), blk, 0, stream>>>(x, xb, nXE);
  gemm_bf16_tdm_kernel<<<gGE, blk, gemmLds, stream>>>(xb, w_saqkv_t + 0 * EE, sa_bqkv + 0 * CE,
                                                      nullptr, qbf, CNT, CE, CE, 0);
  gemm_bf16_tdm_kernel<<<gGE, blk, gemmLds, stream>>>(xb, w_saqkv_t + 1 * EE, sa_bqkv + 1 * CE,
                                                      nullptr, kbf, CNT, CE, CE, 0);
  gemm_bf16_tdm_kernel<<<gGE, blk, gemmLds, stream>>>(xb, w_saqkv_t + 2 * EE, sa_bqkv + 2 * CE,
                                                      vf, nullptr, CNT, CE, CE, 0);
  vtrans_spatial_kernel<<<dim3((unsigned)(nXE / 256)), blk, 0, stream>>>(vf, vtb);
  spatial_attn_kernel<<<dim3(CB * CS * CH), blk, 0, stream>>>(qbf, kbf, vtb, attb);
  gemm_bf16_tdm_kernel<<<gGE, blk, gemmLds, stream>>>(attb, w_sawo_t, sa_bo, proj, nullptr,
                                                      CNT, CE, CE, 0);
  add_ln_kernel<<<dim3(CNT), blk, 0, stream>>>(x, proj, sa_g, sa_b, xA, xb);

  // ================= stage 2: temporal (causal) attention =================
  gemm_bf16_tdm_kernel<<<gGE, blk, gemmLds, stream>>>(xb, w_taqkv_t + 0 * EE, ta_bqkv + 0 * CE,
                                                      nullptr, qbf, CNT, CE, CE, 0);
  gemm_bf16_tdm_kernel<<<gGE, blk, gemmLds, stream>>>(xb, w_taqkv_t + 1 * EE, ta_bqkv + 1 * CE,
                                                      nullptr, kbf, CNT, CE, CE, 0);
  gemm_bf16_tdm_kernel<<<gGE, blk, gemmLds, stream>>>(xb, w_taqkv_t + 2 * EE, ta_bqkv + 2 * CE,
                                                      vf, nullptr, CNT, CE, CE, 0);
  temporal_attn_kernel<<<dim3(CB * CP * CH / 8), blk, 0, stream>>>(qbf, kbf, vf, attb);
  gemm_bf16_tdm_kernel<<<gGE, blk, gemmLds, stream>>>(attb, w_tawo_t, ta_bo, proj, nullptr,
                                                      CNT, CE, CE, 0);
  add_ln_kernel<<<dim3(CNT), blk, 0, stream>>>(xA, proj, ta_g, ta_b, xB, xb);

  // ================= stage 3: FFN =================
  gemm_bf16_tdm_kernel<<<gGF, blk, gemmLds, stream>>>(xb, w_f1_t, f_b1, nullptr, hb,
                                                      CNT, CFF, CE, 1);
  gemm_bf16_tdm_kernel<<<gGE, blk, gemmLds, stream>>>(hb, w_f2_t, f_b2, proj, nullptr,
                                                      CNT, CE, CFF, 0);
  add_ln_kernel<<<dim3(CNT), blk, 0, stream>>>(xB, proj, f_g, f_b, out, nullptr);
}
